// ILMSCAM_Module_38233798869159
// MI455X (gfx1250) — matrix operand order repaired and checked by reading, not compiled, not run on hardware
//
#include <hip/hip_runtime.h>
#include <stdint.h>

// Problem dims (fixed by reference): x [B, C, H, W], N = H*W
#define B_  8
#define C_  256
#define N_  16384
#define KS_ 8            // K-split for GEMM1 parallelism

typedef __attribute__((ext_vector_type(16))) __bf16 v16bf;
typedef __attribute__((ext_vector_type(8)))  float  v8f;

union Frag {
    uint4  q[2];
    v16bf  v;
};

static __device__ __forceinline__ v8f vzero8() {
    v8f z;
#pragma unroll
    for (int i = 0; i < 8; ++i) z[i] = 0.0f;
    return z;
}

static __device__ __forceinline__ unsigned short f32_to_bf16_rne(float f) {
    union { float f; uint32_t u; } t; t.f = f;
    uint32_t u = t.u;
    u += 0x7FFFu + ((u >> 16) & 1u);   // round-to-nearest-even
    return (unsigned short)(u >> 16);
}

// A fragment, 16-bit 16x32 (MxK) per ISA 7.12.2:
// lanes 0-15 : row M=lane,    K = k0+0..7  and k0+16..23
// lanes 16-31: row M=lane-16, K = k0+8..15 and k0+24..31
static __device__ __forceinline__ v16bf load_frag_a(const unsigned short* __restrict__ base,
                                                    int ld, int row0, int k0, int lane) {
    int r  = lane & 15;
    int hi = lane >> 4;
    const unsigned short* p = base + (size_t)(row0 + r) * (size_t)ld + k0 + hi * 8;
    Frag f;
    f.q[0] = *(const uint4*)(p);        // 8 bf16: K chunk 0
    f.q[1] = *(const uint4*)(p + 16);   // 8 bf16: K chunk +16
    return f.v;
}

// B fragment, 16-bit 32x16 (KxN):
// lanes 0-15 : col N=lane,    K = k0+0..7  and k0+16..23
// lanes 16-31: col N=lane-16, K = k0+8..15 and k0+24..31
// `base` is laid out so that column n of B is a contiguous row n of `base`.
static __device__ __forceinline__ v16bf load_frag_b(const unsigned short* __restrict__ base,
                                                    int ld, int col0, int k0, int lane) {
    int c  = lane & 15;
    int hi = lane >> 4;
    const unsigned short* p = base + (size_t)(col0 + c) * (size_t)ld + k0 + hi * 8;
    Frag f;
    f.q[0] = *(const uint4*)(p);
    f.q[1] = *(const uint4*)(p + 16);
    return f.v;
}

static __device__ __forceinline__ v8f wmma_bf16(v16bf a, v16bf b, v8f c) {
    return __builtin_amdgcn_wmma_f32_16x16x32_bf16(false, a, false, b, (short)0, c, false, false);
}

// ---------------------------------------------------------------------------
// Kernel 1: fp32 x -> bf16 q [B,C,N] (row-major) and qT [B,N,C] (transposed)
// 32x32 LDS tile transpose; block (32,8).
// ---------------------------------------------------------------------------
__global__ __launch_bounds__(256)
void k_convert_transpose(const float* __restrict__ x,
                         unsigned short* __restrict__ q,
                         unsigned short* __restrict__ qT) {
    __shared__ float tile[32][33];
    const int n0 = blockIdx.x * 32;
    const int c0 = blockIdx.y * 32;
    const int b  = blockIdx.z;
    const int tx = threadIdx.x, ty = threadIdx.y;

#pragma unroll
    for (int i = ty; i < 32; i += 8) {
        const int c = c0 + i;
        const int n = n0 + tx;
        const size_t o = ((size_t)b * C_ + c) * (size_t)N_ + n;
        const float v = x[o];
        tile[i][tx] = v;
        q[o] = f32_to_bf16_rne(v);
    }
    __syncthreads();
#pragma unroll
    for (int i = ty; i < 32; i += 8) {
        const int n = n0 + i;
        const int c = c0 + tx;
        qT[((size_t)b * N_ + n) * (size_t)C_ + c] = f32_to_bf16_rne(tile[tx][i]);
    }
}

// ---------------------------------------------------------------------------
// Kernel 2: partial energy = q q^T (bf16 WMMA, fp32 acc), symmetric (ti<=tj),
// K split into KS_ chunks. One wave computes a 32x32 tile (2x2 of 16x16).
// part layout: [ks][b][256][256] fp32.
// ---------------------------------------------------------------------------
__global__ __launch_bounds__(32)
void k_energy_partial(const unsigned short* __restrict__ q,
                      float* __restrict__ part) {
    const int lane = threadIdx.x & 31;
    const int b  = blockIdx.z;
    const int ks = blockIdx.y;

    // map blockIdx.x (0..35) -> (ti, tj) with ti <= tj, 8x8 wave-tile grid
    int idx = blockIdx.x, ti = 0;
    while (idx >= 8 - ti) { idx -= 8 - ti; ++ti; }
    const int tj = ti + idx;

    const unsigned short* qb = q + (size_t)b * C_ * N_;
    const int kchunk = N_ / KS_;
    const int kbase  = ks * kchunk;

    v8f acc[2][2];
    acc[0][0] = vzero8(); acc[0][1] = vzero8();
    acc[1][0] = vzero8(); acc[1][1] = vzero8();

    const int i0 = ti * 32, j0 = tj * 32;
    for (int k0 = kbase; k0 < kbase + kchunk; k0 += 32) {
        v16bf a0 = load_frag_a(qb, N_, i0,      k0, lane);
        v16bf a1 = load_frag_a(qb, N_, i0 + 16, k0, lane);
        v16bf b0 = load_frag_b(qb, N_, j0,      k0, lane);
        v16bf b1 = load_frag_b(qb, N_, j0 + 16, k0, lane);
        acc[0][0] = wmma_bf16(a0, b0, acc[0][0]);
        acc[0][1] = wmma_bf16(a0, b1, acc[0][1]);
        acc[1][0] = wmma_bf16(a1, b0, acc[1][0]);
        acc[1][1] = wmma_bf16(a1, b1, acc[1][1]);
    }

    float* pt = part + ((size_t)ks * B_ + b) * (size_t)(C_ * C_);
    const int nloc  = lane & 15;
    const int mbase = (lane >> 4) * 8;    // C/D layout: lanes 16-31 hold M=8..15
#pragma unroll
    for (int si = 0; si < 2; ++si) {
#pragma unroll
        for (int sj = 0; sj < 2; ++sj) {
            const int ib = i0 + si * 16, jb = j0 + sj * 16;
#pragma unroll
            for (int v = 0; v < 8; ++v) {
                const int m = mbase + v;
                const float val = acc[si][sj][v];
                pt[(size_t)(ib + m) * C_ + (jb + nloc)] = val;
                if (ti != tj)  // mirror (energy is symmetric)
                    pt[(size_t)(jb + nloc) * C_ + (ib + m)] = val;
            }
        }
    }
}

// ---------------------------------------------------------------------------
// Kernel 3: reduce K-partials + row softmax of (rowmax - e) == exp(emin-e)/sum.
// One wave per (b,i) row of 256 energies; writes attn as bf16 [B,256,256].
// ---------------------------------------------------------------------------
__global__ __launch_bounds__(32)
void k_softmax(const float* __restrict__ part,
               unsigned short* __restrict__ attn) {
    const int lane = threadIdx.x & 31;
    const int row  = blockIdx.x;          // b*256 + i
    const int b = row >> 8, i = row & 255;

    float e[8];
    float mn = 3.4e38f;
#pragma unroll
    for (int r = 0; r < 8; ++r) {
        const int j = lane + 32 * r;
        const size_t o = (size_t)b * (C_ * C_) + (size_t)i * C_ + j;
        float s = 0.0f;
#pragma unroll
        for (int k = 0; k < KS_; ++k)
            s += part[(size_t)k * B_ * (C_ * C_) + o];
        e[r] = s;
        mn = fminf(mn, s);
    }
#pragma unroll
    for (int off = 16; off > 0; off >>= 1)
        mn = fminf(mn, __shfl_xor(mn, off, 32));

    float p[8];
    float sum = 0.0f;
#pragma unroll
    for (int r = 0; r < 8; ++r) { p[r] = expf(mn - e[r]); sum += p[r]; }
#pragma unroll
    for (int off = 16; off > 0; off >>= 1)
        sum += __shfl_xor(sum, off, 32);
    const float inv = 1.0f / sum;

#pragma unroll
    for (int r = 0; r < 8; ++r) {
        const int j = lane + 32 * r;
        attn[(size_t)b * (C_ * C_) + (size_t)i * C_ + j] = f32_to_bf16_rne(p[r] * inv);
    }
}

// ---------------------------------------------------------------------------
// Kernel 4: out = gamma * (attn @ q) + x, fused epilogue.
// One wave computes a 16(i) x 64(n) tile: A = attn row block (K=d contiguous),
// B columns come from qT rows (d contiguous). K=256 -> 8 WMMA steps x 4 tiles.
// ---------------------------------------------------------------------------
__global__ __launch_bounds__(32)
void k_attn_apply(const unsigned short* __restrict__ attn,
                  const unsigned short* __restrict__ qT,
                  const float* __restrict__ x,
                  const float* __restrict__ gamma,
                  float* __restrict__ out) {
    const int lane = threadIdx.x & 31;
    const int b  = blockIdx.z;
    const int i0 = blockIdx.y * 16;
    const int n0 = blockIdx.x * 64;

    const unsigned short* ab  = attn + (size_t)b * (C_ * C_);
    const unsigned short* qTb = qT   + (size_t)b * N_ * C_;

    v8f acc[4];
#pragma unroll
    for (int t = 0; t < 4; ++t) acc[t] = vzero8();

    for (int d0 = 0; d0 < C_; d0 += 32) {
        v16bf a = load_frag_a(ab, C_, i0, d0, lane);
#pragma unroll
        for (int t = 0; t < 4; ++t) {
            v16bf bf = load_frag_b(qTb, C_, n0 + 16 * t, d0, lane);
            acc[t] = wmma_bf16(a, bf, acc[t]);
        }
    }

    const float g = gamma[0];
    const int nloc  = lane & 15;
    const int mbase = (lane >> 4) * 8;
#pragma unroll
    for (int t = 0; t < 4; ++t) {
#pragma unroll
        for (int v = 0; v < 8; ++v) {
            const int m = mbase + v;
            const size_t o = ((size_t)b * C_ + (i0 + m)) * (size_t)N_ + (n0 + 16 * t + nloc);
            out[o] = g * acc[t][v] + x[o];
        }
    }
}

// ---------------------------------------------------------------------------
extern "C" void kernel_launch(void* const* d_in, const int* in_sizes, int n_in,
                              void* d_out, int out_size, void* d_ws, size_t ws_size,
                              hipStream_t stream) {
    (void)in_sizes; (void)n_in; (void)out_size; (void)ws_size;
    const float* x     = (const float*)d_in[0];
    const float* gamma = (const float*)d_in[1];
    float*       out   = (float*)d_out;

    // Workspace carve-up (~145 MiB):
    //   q_bf16  [B,C,N]      64 MiB
    //   qT_bf16 [B,N,C]      64 MiB
    //   part    [KS,B,C,C]   16 MiB fp32
    //   attn    [B,C,C]       1 MiB bf16
    const size_t nq = (size_t)B_ * C_ * N_;
    unsigned short* q    = (unsigned short*)d_ws;
    unsigned short* qT   = q + nq;
    float*          part = (float*)(qT + nq);
    unsigned short* attn = (unsigned short*)(part + (size_t)KS_ * B_ * C_ * C_);

    k_convert_transpose<<<dim3(N_ / 32, C_ / 32, B_), dim3(32, 8), 0, stream>>>(x, q, qT);
    k_energy_partial<<<dim3(36, KS_, B_), dim3(32), 0, stream>>>(q, part);
    k_softmax<<<dim3(B_ * C_), dim3(32), 0, stream>>>(part, attn);
    k_attn_apply<<<dim3(N_ / 64, C_ / 16, B_), dim3(32), 0, stream>>>(attn, qT, x, gamma, out);
}
